// GPRGNN_3324304687176
// MI455X (gfx1250) — compile-verified, matrix operand order checked by reading, not run on hardware
//
#include <hip/hip_runtime.h>
#include <hip/hip_bf16.h>

// ---------------- problem constants (match reference) ----------------
#define NN   100000   // nodes
#define FIN  512      // input feats
#define HH   64       // hidden
#define CC   16       // classes
#define KHOP 10       // propagation hops
#define KSTEPS (FIN / 32)   // 16 K-steps for GEMM1

typedef __attribute__((ext_vector_type(16))) __bf16        v16bf;
typedef __attribute__((ext_vector_type(8)))  float         v8f;
typedef __attribute__((ext_vector_type(8)))  unsigned short v8us;

// native bf16 convert (backend selects hardware cvt on gfx1250)
__device__ __forceinline__ __bf16 f2bf(float f) { return (__bf16)f; }

// =====================================================================
// Weight pre-pack kernels: lay W1/W2 out in the exact WMMA B-operand
// layout (ISA 7.12.2, 16-bit B 32x16): lane L holds column n=L&15,
// K = kb..kb+7 and 16+kb..16+kb+7 with kb=(L>>4)*8; element i of the
// 16-bf16 lane vector: K = kb+i (i<8), K = 16+kb+(i-8) (i>=8).
// w1p index: ((step*4 + tile)*32 + lane)*16 + i
// =====================================================================
__global__ void k_pack_w1(const float* __restrict__ W1, unsigned short* __restrict__ w1p) {
    int t = blockIdx.x * blockDim.x + threadIdx.x;
    if (t >= KSTEPS * 4 * 32 * 16) return;
    int i    = t & 15;
    int lane = (t >> 4) & 31;
    int tile = (t >> 9) & 3;
    int step = t >> 11;
    int n  = lane & 15;
    int kb = (lane >> 4) << 3;
    int K  = step * 32 + ((i < 8) ? (kb + i) : (16 + kb + (i - 8)));
    unsigned short bits = __builtin_bit_cast(unsigned short, f2bf(W1[(size_t)K * HH + tile * 16 + n]));
    w1p[t] = bits;
}

__global__ void k_pack_w2(const float* __restrict__ W2, unsigned short* __restrict__ w2p) {
    int t = blockIdx.x * blockDim.x + threadIdx.x;
    if (t >= 2 * 32 * 16) return;
    int i    = t & 15;
    int lane = (t >> 4) & 31;
    int step = t >> 9;
    int n  = lane & 15;
    int kb = (lane >> 4) << 3;
    int K  = step * 32 + ((i < 8) ? (kb + i) : (16 + kb + (i - 8)));
    w2p[t] = __builtin_bit_cast(unsigned short, f2bf(W2[K * CC + n]));
}

// =====================================================================
// Fused MLP: hA = relu(x@W1+b1) @ W2 + b2   [NN x CC] f32
// Block = 64 threads = 2 waves; each wave owns a private 16-row slab and
// computes ALL four 16x16 GEMM1 col tiles (A loaded once -> 4 WMMAs per
// K-step), then its own GEMM2 from its private LDS region.
// =====================================================================
__global__ __launch_bounds__(64) void k_mlp_fused(
    const float* __restrict__ x,
    const unsigned short* __restrict__ w1p, const float* __restrict__ b1,
    const unsigned short* __restrict__ w2p, const float* __restrict__ b2,
    float* __restrict__ hA)
{
    __shared__ unsigned short lds[2 * 16 * HH];   // per-wave bf16 h1 slabs

    const int wave = threadIdx.x >> 5;
    const int lane = threadIdx.x & 31;
    const int m    = lane & 15;
    const int kb   = (lane >> 4) << 3;            // 0 or 8
    const int r0   = blockIdx.x * 32 + wave * 16; // this wave's 16 rows

    const v16bf* __restrict__ B1 = (const v16bf*)w1p;
    const v16bf* __restrict__ B2 = (const v16bf*)w2p;

    // ---------------- GEMM1: 4 col tiles, K-loop over 512 ----------------
    v8f acc[4] = {v8f{}, v8f{}, v8f{}, v8f{}};
    const float* __restrict__ xrow = x + (size_t)(r0 + m) * FIN;

    for (int step = 0; step < KSTEPS; ++step) {
        const int k0 = step << 5;
        float4 a0 = *(const float4*)(xrow + k0 + kb);
        float4 a1 = *(const float4*)(xrow + k0 + kb + 4);
        float4 a2 = *(const float4*)(xrow + k0 + 16 + kb);
        float4 a3 = *(const float4*)(xrow + k0 + 16 + kb + 4);
        float af[16] = { a0.x,a0.y,a0.z,a0.w, a1.x,a1.y,a1.z,a1.w,
                         a2.x,a2.y,a2.z,a2.w, a3.x,a3.y,a3.z,a3.w };
        v16bf A;
        #pragma unroll
        for (int i = 0; i < 16; ++i) A[i] = f2bf(af[i]);

        #pragma unroll
        for (int tile = 0; tile < 4; ++tile) {
            v16bf Bt = B1[(size_t)((step << 2) | tile) * 32 + lane];
            acc[tile] = __builtin_amdgcn_wmma_f32_16x16x32_bf16(
                            false, A, false, Bt, (short)0, acc[tile], false, false);
        }
    }

    // ---------------- epilogue: bias + ReLU -> private LDS slab ----------------
    const int mo   = (lane >> 4) << 3;            // C layout: M = mo + r
    const int base = wave * 16 * HH;
    #pragma unroll
    for (int tile = 0; tile < 4; ++tile) {
        const float bias1 = b1[tile * 16 + m];
        #pragma unroll
        for (int r = 0; r < 8; ++r) {
            float v = acc[tile][r] + bias1;
            v = v > 0.0f ? v : 0.0f;
            lds[base + (mo + r) * HH + tile * 16 + m] =
                __builtin_bit_cast(unsigned short, f2bf(v));
        }
    }
    __syncthreads();

    // ---------------- GEMM2: 16x64 @ 64x16 from private LDS ----------------
    v8f acc2 = {};
    #pragma unroll
    for (int step = 0; step < 2; ++step) {
        const int ks = step << 5;
        // A2: two 16B ds_load_b128 per lane (row m, K runs of 8)
        v8us lo = *(const v8us*)&lds[base + m * HH + ks + kb];
        v8us hi = *(const v8us*)&lds[base + m * HH + ks + 16 + kb];
        union { v16bf v; v8us u[2]; } a2;
        a2.u[0] = lo; a2.u[1] = hi;
        v16bf Bt = B2[(size_t)step * 32 + lane];
        acc2 = __builtin_amdgcn_wmma_f32_16x16x32_bf16(
                   false, a2.v, false, Bt, (short)0, acc2, false, false);
    }
    const float bias2 = b2[m];
    #pragma unroll
    for (int r = 0; r < 8; ++r)
        hA[(size_t)(r0 + mo + r) * CC + m] = acc2[r] + bias2;
}

// =====================================================================
// Riemann normalize hA in place; hidden = temp[0] * h_norm
// =====================================================================
__global__ __launch_bounds__(256) void k_riemann_init(
    float* __restrict__ hA, const float* __restrict__ wfn,
    const float* __restrict__ temp, float* __restrict__ hidden)
{
    int i = blockIdx.x * blockDim.x + threadIdx.x;
    if (i >= NN) return;
    float4* row = (float4*)(hA + (size_t)i * CC);
    float4 v[4] = { row[0], row[1], row[2], row[3] };
    float h[16];
    #pragma unroll
    for (int q = 0; q < 4; ++q) {
        h[4*q+0] = v[q].x; h[4*q+1] = v[q].y; h[4*q+2] = v[q].z; h[4*q+3] = v[q].w;
    }
    float s = 0.01f;
    #pragma unroll
    for (int c = 0; c < 16; ++c) s += h[c] * h[c] * fabsf(wfn[c]);
    float inv = 1.0f / (sqrtf(s) + 0.01f);
    float t0 = temp[0];
    float4* out = (float4*)(hidden + (size_t)i * CC);
    #pragma unroll
    for (int q = 0; q < 4; ++q) {
        float4 hn = make_float4(h[4*q+0]*inv, h[4*q+1]*inv, h[4*q+2]*inv, h[4*q+3]*inv);
        row[q] = hn;
        out[q] = make_float4(t0*hn.x, t0*hn.y, t0*hn.z, t0*hn.w);
    }
}

// ---------------- degree / dis ----------------
__global__ void k_deg_init(float* __restrict__ deg) {
    int i = blockIdx.x * blockDim.x + threadIdx.x;
    if (i < NN) deg[i] = 1.0f;                       // self-loop
}
__global__ void k_deg_accum(const int* __restrict__ cols,
                            float* __restrict__ deg, int E) {
    int e = blockIdx.x * blockDim.x + threadIdx.x;
    if (e < E) atomicAdd(&deg[cols[e]], 1.0f);
}
__global__ void k_dis(float* __restrict__ deg) {
    int i = blockIdx.x * blockDim.x + threadIdx.x;
    if (i < NN) deg[i] = rsqrtf(deg[i]);             // deg >= 1 always
}

// ---------------- per-hop kernels ----------------
__global__ void k_zero(float* __restrict__ hB) {
    int t = blockIdx.x * blockDim.x + threadIdx.x;
    if (t < NN * CC) hB[t] = 0.0f;
}

// one lane per (edge, 4 channels): b128 gather, scale, 4 atomic f32 scatters
__global__ __launch_bounds__(256) void k_scatter(
    const int* __restrict__ rows, const int* __restrict__ cols,
    const float* __restrict__ dis, const float* __restrict__ hA,
    float* __restrict__ hB, int E)
{
    long long t = (long long)blockIdx.x * blockDim.x + threadIdx.x;
    if (t >= (long long)E * 4) return;
    int e = (int)(t >> 2);
    int q = (int)(t & 3);
    int r  = rows[e];
    int co = cols[e];
    float nv = dis[r] * dis[co];
    float4 hv = *(const float4*)(hA + (size_t)r * CC + q * 4);
    float* dst = hB + (size_t)co * CC + q * 4;
    atomicAdd(dst + 0, nv * hv.x);
    atomicAdd(dst + 1, nv * hv.y);
    atomicAdd(dst + 2, nv * hv.z);
    atomicAdd(dst + 3, nv * hv.w);
}

// fold self-loop (dis^2 * h_old) into hB, Riemann-normalize, write hA,
// hidden += temp[k+1] * h_new
__global__ __launch_bounds__(256) void k_riemann_step(
    const float* __restrict__ hB, float* __restrict__ hA,
    const float* __restrict__ dis, const float* __restrict__ wfn,
    const float* __restrict__ temp, int k, float* __restrict__ hidden)
{
    int i = blockIdx.x * blockDim.x + threadIdx.x;
    if (i >= NN) return;
    float d2 = dis[i] * dis[i];
    const float4* bin = (const float4*)(hB + (size_t)i * CC);
    float4*       ain = (float4*)(hA + (size_t)i * CC);
    float h[16];
    #pragma unroll
    for (int q = 0; q < 4; ++q) {
        float4 vb = bin[q];
        float4 va = ain[q];
        h[4*q+0] = vb.x + d2 * va.x;  h[4*q+1] = vb.y + d2 * va.y;
        h[4*q+2] = vb.z + d2 * va.z;  h[4*q+3] = vb.w + d2 * va.w;
    }
    float s = 0.01f;
    #pragma unroll
    for (int c = 0; c < 16; ++c) s += h[c] * h[c] * fabsf(wfn[c]);
    float inv = 1.0f / (sqrtf(s) + 0.01f);
    float tk = temp[k + 1];
    float4* out = (float4*)(hidden + (size_t)i * CC);
    #pragma unroll
    for (int q = 0; q < 4; ++q) {
        float4 hn = make_float4(h[4*q+0]*inv, h[4*q+1]*inv, h[4*q+2]*inv, h[4*q+3]*inv);
        ain[q] = hn;
        float4 o = out[q];
        out[q] = make_float4(o.x + tk*hn.x, o.y + tk*hn.y, o.z + tk*hn.z, o.w + tk*hn.w);
    }
}

// =====================================================================
extern "C" void kernel_launch(void* const* d_in, const int* in_sizes, int n_in,
                              void* d_out, int out_size, void* d_ws, size_t ws_size,
                              hipStream_t stream)
{
    const float* x    = (const float*)d_in[0];
    const int*   ei   = (const int*)  d_in[1];
    const float* W1   = (const float*)d_in[2];
    const float* b1   = (const float*)d_in[3];
    const float* W2   = (const float*)d_in[4];
    const float* b2   = (const float*)d_in[5];
    const float* temp = (const float*)d_in[6];
    const float* wfn  = (const float*)d_in[7];

    const int E = in_sizes[1] / 2;
    const int* rows = ei;          // edge_index[0] = source
    const int* cols = ei + E;      // edge_index[1] = destination

    float* hA  = (float*)d_ws;                          // NN*CC f32
    float* hB  = hA + (size_t)NN * CC;                  // NN*CC f32
    float* deg = hB + (size_t)NN * CC;                  // NN f32 (becomes dis)
    unsigned short* w1p = (unsigned short*)(deg + NN);  // 32768 bf16 (32B-aligned)
    unsigned short* w2p = w1p + KSTEPS * 4 * 32 * 16;   // 1024 bf16
    float* hidden = (float*)d_out;                      // NN*CC f32

    // pre-pack weights into WMMA B-operand bf16 layout
    k_pack_w1<<<(KSTEPS * 4 * 32 * 16 + 255) / 256, 256, 0, stream>>>(W1, w1p);
    k_pack_w2<<<(2 * 32 * 16 + 255) / 256, 256, 0, stream>>>(W2, w2p);

    // MLP (fused GEMM1+ReLU+GEMM2, WMMA bf16)
    k_mlp_fused<<<NN / 32, 64, 0, stream>>>(x, w1p, b1, w2p, b2, hA);

    // first Riemann + hidden init
    k_riemann_init<<<(NN + 255) / 256, 256, 0, stream>>>(hA, wfn, temp, hidden);

    // symmetric gcn_norm degrees (self-loops included)
    k_deg_init <<<(NN + 255) / 256, 256, 0, stream>>>(deg);
    k_deg_accum<<<(E  + 255) / 256, 256, 0, stream>>>(cols, deg, E);
    k_dis      <<<(NN + 255) / 256, 256, 0, stream>>>(deg);

    // K propagation hops
    long long scat_threads = (long long)E * 4;
    int scat_blocks = (int)((scat_threads + 255) / 256);
    for (int k = 0; k < KHOP; ++k) {
        k_zero        <<<(NN * CC + 255) / 256, 256, 0, stream>>>(hB);
        k_scatter     <<<scat_blocks, 256, 0, stream>>>(rows, cols, deg, hA, hB, E);
        k_riemann_step<<<(NN + 255) / 256, 256, 0, stream>>>(hB, hA, deg, wfn, temp, k, hidden);
    }
}
